// FastRCNN_29858612641848
// MI455X (gfx1250) — matrix operand order repaired and checked by reading, not compiled, not run on hardware
//
#include <hip/hip_runtime.h>
#include <hip/hip_bf16.h>

typedef __attribute__((ext_vector_type(16))) _Float16 v16h;
typedef __attribute__((ext_vector_type(8)))  float    v8f;

#define ROI_OUT 7
#define POOL_C 512
#define POOL_H 50
#define POOL_W 50
#define PD     (POOL_C * ROI_OUT * ROI_OUT)   // 25088
#define NROI   512
#define DDIM   4096
#define DHEAD  1024
#define NHEAD  4

#define BM 128
#define BN 128
#define BK 32
#define LDSK 40   // 32 + 8 halfs pad -> 80B row stride, 16B-aligned rows

// ---------------------------------------------------------------------------
// ROI max pooling: one block per ROI, threads sweep (channel, bin) pairs.
// ---------------------------------------------------------------------------
__global__ void roipool_kernel(const float* __restrict__ x,
                               const float* __restrict__ rois,
                               const int*   __restrict__ roi_idx,
                               float*       __restrict__ pool) {
  const int n = blockIdx.x;
  const float y1 = rois[n * 4 + 0], x1 = rois[n * 4 + 1];
  const float y2 = rois[n * 4 + 2], x2 = rois[n * 4 + 3];
  const int xmn = (int)rintf(x1 * 0.0625f);
  const int ymn = (int)rintf(y1 * 0.0625f);
  const int xmx = (int)rintf(x2 * 0.0625f);
  const int ymx = (int)rintf(y2 * 0.0625f);
  const float rw = (float)max(xmx - xmn + 1, 1);
  const float rh = (float)max(ymx - ymn + 1, 1);
  int xs[7], xe[7], ys[7], ye[7];
#pragma unroll
  for (int i = 0; i < 7; ++i) {
    xs[i] = xmn + (int)floorf((float)i * rw / 7.0f);
    xe[i] = xmn + (int)ceilf((float)(i + 1) * rw / 7.0f);
    ys[i] = ymn + (int)floorf((float)i * rh / 7.0f);
    ye[i] = ymn + (int)ceilf((float)(i + 1) * rh / 7.0f);
  }
  const int bi = roi_idx[n];
  const float* feat = x + (size_t)bi * POOL_C * POOL_H * POOL_W;
  for (int idx = threadIdx.x; idx < POOL_C * 49; idx += blockDim.x) {
    const int c = idx / 49, bin = idx % 49;
    const int by = bin / 7, bx = bin % 7;
    const int x0 = max(xs[bx], 0), xF = min(xe[bx], POOL_W);
    const int y0 = max(ys[by], 0), yF = min(ye[by], POOL_H);
    float m = -1e30f;
    const float* fc = feat + (size_t)c * POOL_H * POOL_W;
    for (int yy = y0; yy < yF; ++yy)
      for (int xx = x0; xx < xF; ++xx)
        m = fmaxf(m, fc[yy * POOL_W + xx]);
    pool[(size_t)n * PD + idx] = (m <= -5e29f) ? 0.0f : m;
  }
}

// ---------------------------------------------------------------------------
// Geometric gate: g[i][j] = relu(tg . w + b) + 1 over all 512x512 pairs.
// ---------------------------------------------------------------------------
__global__ void gate_kernel(const float* __restrict__ rois,
                            const float* __restrict__ w,
                            const float* __restrict__ b,
                            float* __restrict__ g) {
  const int t = blockIdx.x * blockDim.x + threadIdx.x;
  if (t >= NROI * NROI) return;
  const int i = t / NROI, j = t % NROI;
  const float xmn_i = rois[i * 4 + 1], xmx_i = rois[i * 4 + 3];
  const float ymn_i = rois[i * 4 + 0], ymx_i = rois[i * 4 + 2];
  const float xmn_j = rois[j * 4 + 1], xmx_j = rois[j * 4 + 3];
  const float ymn_j = rois[j * 4 + 0], ymx_j = rois[j * 4 + 2];
  const float cxi = 0.5f * (xmn_i + xmx_i), cyi = 0.5f * (ymn_i + ymx_i);
  const float cxj = 0.5f * (xmn_j + xmx_j), cyj = 0.5f * (ymn_j + ymx_j);
  const float hwi = fabsf(cxi - xmx_i), hhi = fabsf(cyi - ymx_i);
  const float hwj = fabsf(cxj - xmx_j), hhj = fabsf(cyj - ymx_j);
  const float tx = fmaxf(fabsf(cxi - cxj), 1e-3f) / hwi;
  const float ty = fmaxf(fabsf(cyi - cyj), 1e-3f) / hhi;
  const float tw = hwi / hwj, th = hhi / hhj;
  const float v = tx * w[0] + ty * w[1] + tw * w[2] + th * w[3] + b[0];
  g[t] = fmaxf(v, 0.0f) + 1.0f;
}

// ---------------------------------------------------------------------------
// Tiled WMMA GEMM:  C = relu?( alpha * A@B(^T) + bias )
// fp32 operands converted to f16 tiles in LDS; f32 accumulation via
// v_wmma_f32_16x16x32_f16. 256 threads = 8 wave32s, wave = 64x32 patch.
// Double-buffered LDS (one barrier per K-tile); next-tile global_load_b128s
// issue before the barrier so VMEM latency hides under ds_load+wmma.
// Batched via blockIdx.z with element strides. All dims multiples of tiles.
// ---------------------------------------------------------------------------
template <bool TRANS_B, bool RELU>
__global__ __launch_bounds__(256)
void gemm_wmma(const float* __restrict__ A, const float* __restrict__ B,
               const float* __restrict__ bias, float* __restrict__ C,
               int M, int N, int K, int ldc,
               long long sA, long long sB, long long sBias, long long sC,
               float alpha) {
  __shared__ alignas(16) _Float16 As[2 * BM * LDSK];
  __shared__ alignas(16) _Float16 Bs[2 * BN * LDSK];

  const int z = blockIdx.z;
  A += (size_t)z * sA;
  B += (size_t)z * sB;
  C += (size_t)z * sC;
  if (bias) bias += (size_t)z * sBias;

  const int bn0 = blockIdx.x * BN;
  const int bm0 = blockIdx.y * BM;
  const int tid = threadIdx.x;
  const int lane = tid & 31;
  const int wid = tid >> 5;
  const int wm = (wid & 1) * 64;   // wave row offset in block tile
  const int wn = (wid >> 1) * 32;  // wave col offset in block tile
  const int lm = lane & 15;        // row/col within 16x16 fragment
  const int kh = lane >> 4;        // K-half select per ISA 16-bit layouts

  v8f acc[4][2] = {};

  // loader assignments
  const int lr = tid >> 1;        // 0..127 row (A / NT-B)
  const int lk = (tid & 1) * 16;  // 0 or 16 K chunk
  const int bkg = tid >> 5;       // 0..7   NN-B K-group (4 rows)
  const int bng = tid & 31;       // 0..31  NN-B N-group (4 cols)

  float4 stA[4], stB[4];

  auto load_global = [&](int k0) {
    const float* srcA = A + (size_t)(bm0 + lr) * (size_t)K + k0 + lk;
#pragma unroll
    for (int u = 0; u < 4; ++u) stA[u] = ((const float4*)srcA)[u];
    if (TRANS_B) {
      const float* srcB = B + (size_t)(bn0 + lr) * (size_t)K + k0 + lk;
#pragma unroll
      for (int u = 0; u < 4; ++u) stB[u] = ((const float4*)srcB)[u];
    } else {
      // 4 consecutive K rows, 4 consecutive N cols per thread (coalesced)
      const float* srcB =
          B + (size_t)(k0 + bkg * 4) * (size_t)N + bn0 + bng * 4;
#pragma unroll
      for (int r = 0; r < 4; ++r)
        stB[r] = *(const float4*)(srcB + (size_t)r * N);
    }
  };

  auto store_lds = [&](int buf) {
    _Float16* Asb = &As[buf * BM * LDSK];
    _Float16* Bsb = &Bs[buf * BN * LDSK];
    {
      alignas(16) _Float16 tmp[16];
#pragma unroll
      for (int u = 0; u < 4; ++u) {
        tmp[4 * u + 0] = (_Float16)stA[u].x;
        tmp[4 * u + 1] = (_Float16)stA[u].y;
        tmp[4 * u + 2] = (_Float16)stA[u].z;
        tmp[4 * u + 3] = (_Float16)stA[u].w;
      }
      *(uint4*)&Asb[lr * LDSK + lk]     = *(const uint4*)&tmp[0];
      *(uint4*)&Asb[lr * LDSK + lk + 8] = *(const uint4*)&tmp[8];
    }
    if (TRANS_B) {
      alignas(16) _Float16 tmp[16];
#pragma unroll
      for (int u = 0; u < 4; ++u) {
        tmp[4 * u + 0] = (_Float16)stB[u].x;
        tmp[4 * u + 1] = (_Float16)stB[u].y;
        tmp[4 * u + 2] = (_Float16)stB[u].z;
        tmp[4 * u + 3] = (_Float16)stB[u].w;
      }
      *(uint4*)&Bsb[lr * LDSK + lk]     = *(const uint4*)&tmp[0];
      *(uint4*)&Bsb[lr * LDSK + lk + 8] = *(const uint4*)&tmp[8];
    } else {
      // transpose 4x4 micro-tile: per N column pack 4 K-halfs -> ds_store_b64
#pragma unroll
      for (int c = 0; c < 4; ++c) {
        alignas(8) _Float16 col[4];
#pragma unroll
        for (int r = 0; r < 4; ++r)
          col[r] = (_Float16)((const float*)&stB[r])[c];
        *(unsigned long long*)&Bsb[(bng * 4 + c) * LDSK + bkg * 4] =
            *(const unsigned long long*)col;
      }
    }
  };

  // prologue: stage first tile
  load_global(0);
  store_lds(0);

  int cur = 0;
  for (int k0 = 0; k0 < K; k0 += BK) {
    const bool has_next = (k0 + BK) < K;
    if (has_next) load_global(k0 + BK);  // issue next-tile VMEM early
    if (k0 + 2 * BK < K)                 // warm L2 two tiles ahead
      __builtin_prefetch(A + (size_t)(bm0 + lr) * (size_t)K + k0 + 2 * BK + lk,
                         0, 1);
    __syncthreads();  // buf `cur` fully stored; prior reads of buf cur^1 done

    const _Float16* Asb = &As[cur * BM * LDSK];
    const _Float16* Bsb = &Bs[cur * BN * LDSK];
    // fragments per ISA 16-bit A(16x32)/B(32x16) lane layouts
    v16h afrag[4], bfrag[2];
#pragma unroll
    for (int mt = 0; mt < 4; ++mt) {
      const _Float16* ap = &Asb[(wm + mt * 16 + lm) * LDSK];
      union { uint4 u[2]; v16h v; } t;
      t.u[0] = *(const uint4*)&ap[kh * 8];       // K = kh*8 .. +7
      t.u[1] = *(const uint4*)&ap[16 + kh * 8];  // K = 16+kh*8 .. +7
      afrag[mt] = t.v;
    }
#pragma unroll
    for (int nt = 0; nt < 2; ++nt) {
      const _Float16* bp = &Bsb[(wn + nt * 16 + lm) * LDSK];
      union { uint4 u[2]; v16h v; } t;
      t.u[0] = *(const uint4*)&bp[kh * 8];       // K = kh*8 .. +7
      t.u[1] = *(const uint4*)&bp[16 + kh * 8];  // K = 16+kh*8 .. +7
      bfrag[nt] = t.v;
    }
#pragma unroll
    for (int mt = 0; mt < 4; ++mt)
#pragma unroll
      for (int nt = 0; nt < 2; ++nt)
        acc[mt][nt] = __builtin_amdgcn_wmma_f32_16x16x32_f16(
            false, afrag[mt], false, bfrag[nt], (short)0, acc[mt][nt],
            false, false);

    if (has_next) store_lds(cur ^ 1);
    cur ^= 1;
  }

  // epilogue per 32-bit C/D layout: VGPR r -> row kh*8+r, lane&15 -> col
#pragma unroll
  for (int mt = 0; mt < 4; ++mt) {
    const int row0 = bm0 + wm + mt * 16 + kh * 8;
#pragma unroll
    for (int nt = 0; nt < 2; ++nt) {
      const int c = bn0 + wn + nt * 16 + lm;
      const float bv = bias ? bias[c] : 0.0f;
#pragma unroll
      for (int r = 0; r < 8; ++r) {
        float v = acc[mt][nt][r] * alpha + bv;
        if (RELU) v = fmaxf(v, 0.0f);
        C[(size_t)(row0 + r) * (size_t)ldc + c] = v;
      }
    }
  }
}

// ---------------------------------------------------------------------------
// Row softmax over k with geo gate: in pwaT[h][q][k], out w[h][q][k].
// ---------------------------------------------------------------------------
__global__ void softmax_gate_kernel(const float* __restrict__ pwaT,
                                    const float* __restrict__ g,
                                    float* __restrict__ w) {
  const int q = blockIdx.x, h = blockIdx.y;
  const float* row = pwaT + ((size_t)h * NROI + q) * NROI;
  float* out = w + ((size_t)h * NROI + q) * NROI;
  __shared__ float red[256];
  float m = -1e30f;
  for (int k = threadIdx.x; k < NROI; k += 256) m = fmaxf(m, row[k]);
  red[threadIdx.x] = m;
  __syncthreads();
  for (int s = 128; s > 0; s >>= 1) {
    if (threadIdx.x < s)
      red[threadIdx.x] = fmaxf(red[threadIdx.x], red[threadIdx.x + s]);
    __syncthreads();
  }
  m = red[0];
  __syncthreads();
  float sum = 0.0f;
  for (int k = threadIdx.x; k < NROI; k += 256) {
    const float e = __expf(row[k] - m) * g[(size_t)k * NROI + q];
    out[k] = e;
    sum += e;
  }
  red[threadIdx.x] = sum;
  __syncthreads();
  for (int s = 128; s > 0; s >>= 1) {
    if (threadIdx.x < s) red[threadIdx.x] += red[threadIdx.x + s];
    __syncthreads();
  }
  const float inv = 1.0f / red[0];
  for (int k = threadIdx.x; k < NROI; k += 256) out[k] *= inv;
}

__global__ void add_kernel(const float* __restrict__ a,
                           const float* __restrict__ b,
                           float* __restrict__ c, int n) {
  const int i = blockIdx.x * blockDim.x + threadIdx.x;
  if (i < n) c[i] = a[i] + b[i];
}

// ---------------------------------------------------------------------------
// Tiny loc/score heads: one block per ROI, threads own output columns
// (coalesced weight reads, fc row broadcast through L2).
// ---------------------------------------------------------------------------
__global__ void heads_kernel(const float* __restrict__ fc,
                             const float* __restrict__ loc_w,
                             const float* __restrict__ loc_b,
                             const float* __restrict__ score_w,
                             const float* __restrict__ score_b,
                             float* __restrict__ out) {
  const int row = blockIdx.x;
  const float* f = fc + (size_t)row * DDIM;
  for (int o = threadIdx.x; o < 84 + 21; o += blockDim.x) {
    if (o < 84) {
      float s = loc_b[o];
      for (int k = 0; k < DDIM; ++k) s += f[k] * loc_w[(size_t)k * 84 + o];
      out[(size_t)row * 84 + o] = s;
    } else {
      const int oo = o - 84;
      float s = score_b[oo];
      for (int k = 0; k < DDIM; ++k) s += f[k] * score_w[(size_t)k * 21 + oo];
      out[(size_t)NROI * 84 + (size_t)row * 21 + oo] = s;
    }
  }
}

// ---------------------------------------------------------------------------
extern "C" void kernel_launch(void* const* d_in, const int* in_sizes, int n_in,
                              void* d_out, int out_size, void* d_ws,
                              size_t ws_size, hipStream_t stream) {
  (void)in_sizes; (void)n_in; (void)out_size; (void)ws_size;
  const float* x       = (const float*)d_in[0];
  const float* rois    = (const float*)d_in[1];
  const int*   roiidx  = (const int*)d_in[2];
  const float* fc7_w   = (const float*)d_in[3];
  const float* fc7_b   = (const float*)d_in[4];
  const float* fc8_w   = (const float*)d_in[5];
  const float* fc8_b   = (const float*)d_in[6];
  const float* wg1_w   = (const float*)d_in[7];
  const float* wg1_b   = (const float*)d_in[8];
  const float* wg2_w   = (const float*)d_in[9];
  const float* wg2_b   = (const float*)d_in[10];
  const float* Wk1_w = (const float*)d_in[11]; const float* Wk1_b = (const float*)d_in[12];
  const float* Wq1_w = (const float*)d_in[13]; const float* Wq1_b = (const float*)d_in[14];
  const float* Wv1_w = (const float*)d_in[15]; const float* Wv1_b = (const float*)d_in[16];
  const float* Wk2_w = (const float*)d_in[17]; const float* Wk2_b = (const float*)d_in[18];
  const float* Wq2_w = (const float*)d_in[19]; const float* Wq2_b = (const float*)d_in[20];
  const float* Wv2_w = (const float*)d_in[21]; const float* Wv2_b = (const float*)d_in[22];
  const float* cls_w = (const float*)d_in[23]; const float* cls_b = (const float*)d_in[24];
  const float* loc_w = (const float*)d_in[25]; const float* loc_b = (const float*)d_in[26];
  const float* score_w = (const float*)d_in[27]; const float* score_b = (const float*)d_in[28];
  float* out = (float*)d_out;

  // workspace layout (floats)
  float* ws    = (float*)d_ws;
  float* pool  = ws;                    // 512*25088
  float* fc7b  = pool + (size_t)NROI * PD;
  float* fc8b  = fc7b + (size_t)NROI * DDIM;
  float* tbuf  = fc8b + (size_t)NROI * DDIM;
  float* clsb  = tbuf + (size_t)NROI * DDIM;
  float* Kb    = clsb + (size_t)NROI * DDIM;      // 4*512*1024
  float* Qb    = Kb + (size_t)NHEAD * NROI * DHEAD;
  float* Vb    = Qb + (size_t)NHEAD * NROI * DHEAD;
  float* relb  = Vb + (size_t)NHEAD * NROI * DHEAD;  // 512*4096
  float* pwa   = relb + (size_t)NROI * DDIM;         // 4*512*512
  float* wat   = pwa + (size_t)NHEAD * NROI * NROI;  // 4*512*512
  float* g1    = wat + (size_t)NHEAD * NROI * NROI;  // 512*512
  float* g2    = g1 + (size_t)NROI * NROI;

  const long long sW   = (long long)DDIM * DHEAD;  // 4096*1024
  const long long sKQV = (long long)NROI * DHEAD;  // 512*1024
  const long long sP   = (long long)NROI * NROI;   // 512*512

  // 1) ROI pool + geometric gates
  roipool_kernel<<<NROI, 256, 0, stream>>>(x, rois, roiidx, pool);
  gate_kernel<<<(NROI * NROI) / 256, 256, 0, stream>>>(rois, wg1_w, wg1_b, g1);
  gate_kernel<<<(NROI * NROI) / 256, 256, 0, stream>>>(rois, wg2_w, wg2_b, g2);

  // 2) fc7 = relu(pool @ fc7_w + b)    (512 x 25088) @ (25088 x 4096)
  gemm_wmma<false, true><<<dim3(DDIM / BN, NROI / BM, 1), 256, 0, stream>>>(
      pool, fc7_w, fc7_b, fc7b, NROI, DDIM, PD, DDIM, 0, 0, 0, 0, 1.0f);

  auto relation = [&](const float* feat, const float* kw, const float* kb,
                      const float* qw, const float* qb, const float* vw,
                      const float* vb, const float* g, float* relout) {
    // K/Q/V: batched over 4 heads, (512 x 4096) @ (4096 x 1024)
    gemm_wmma<false, false><<<dim3(DHEAD / BN, NROI / BM, NHEAD), 256, 0, stream>>>(
        feat, kw, kb, Kb, NROI, DHEAD, DDIM, DHEAD, 0, sW, DHEAD, sKQV, 1.0f);
    gemm_wmma<false, false><<<dim3(DHEAD / BN, NROI / BM, NHEAD), 256, 0, stream>>>(
        feat, qw, qb, Qb, NROI, DHEAD, DDIM, DHEAD, 0, sW, DHEAD, sKQV, 1.0f);
    gemm_wmma<false, false><<<dim3(DHEAD / BN, NROI / BM, NHEAD), 256, 0, stream>>>(
        feat, vw, vb, Vb, NROI, DHEAD, DDIM, DHEAD, 0, sW, DHEAD, sKQV, 1.0f);
    // pwaT[h][q][k] = (Q . K) / 32  -> NT gemm, (512 x 1024) x (512 x 1024)^T
    gemm_wmma<true, false><<<dim3(NROI / BN, NROI / BM, NHEAD), 256, 0, stream>>>(
        Qb, Kb, nullptr, pwa, NROI, NROI, DHEAD, NROI, sKQV, sKQV, 0, sP,
        1.0f / 32.0f);
    // softmax over k, gated by g[k][q]
    softmax_gate_kernel<<<dim3(NROI, NHEAD), 256, 0, stream>>>(pwa, g, wat);
    // rel[q][h*1024+d] = sum_k w[h][q][k] * V[h][k][d]  -> NN gemm into
    // head-interleaved output (ldc=4096, per-head col offset via strideC)
    gemm_wmma<false, false><<<dim3(DHEAD / BN, NROI / BM, NHEAD), 256, 0, stream>>>(
        wat, Vb, nullptr, relout, NROI, DHEAD, NROI, DDIM, sP, sKQV, 0,
        (long long)DHEAD, 1.0f);
  };

  // 3) relation 1, fc8
  relation(fc7b, Wk1_w, Wk1_b, Wq1_w, Wq1_b, Wv1_w, Wv1_b, g1, relb);
  add_kernel<<<(NROI * DDIM) / 256, 256, 0, stream>>>(fc7b, relb, tbuf,
                                                      NROI * DDIM);
  gemm_wmma<false, true><<<dim3(DDIM / BN, NROI / BM, 1), 256, 0, stream>>>(
      tbuf, fc8_w, fc8_b, fc8b, NROI, DDIM, DDIM, DDIM, 0, 0, 0, 0, 1.0f);

  // 4) relation 2, cls
  relation(fc8b, Wk2_w, Wk2_b, Wq2_w, Wq2_b, Wv2_w, Wv2_b, g2, relb);
  add_kernel<<<(NROI * DDIM) / 256, 256, 0, stream>>>(fc8b, relb, tbuf,
                                                      NROI * DDIM);
  gemm_wmma<false, false><<<dim3(DDIM / BN, NROI / BM, 1), 256, 0, stream>>>(
      tbuf, cls_w, cls_b, clsb, NROI, DDIM, DDIM, DDIM, 0, 0, 0, 0, 1.0f);

  // 5) loc + score heads -> d_out (locs flat, then scores flat)
  heads_kernel<<<NROI, 128, 0, stream>>>(clsb, loc_w, loc_b, score_w, score_b,
                                         out);
}
